// BidirectionalAttention_8194797601111
// MI455X (gfx1250) — matrix operand order repaired and checked by reading, not compiled, not run on hardware
//
#include <hip/hip_runtime.h>

// MI455X / gfx1250, wave32. Attention block + LIF readout.
// All matrix math via v_wmma_f32_16x16x32_f16. GEMM and attention K/V tiles
// staged through double-buffered LDS by the Tensor Data Mover
// (tensor_load_to_lds + s_wait_tensorcnt), fragments read with ds_load_b128.
// Cross-lane reductions use immediate-pattern ds_swizzle.
// Workspace requirement: ~185 MB.

constexpr int NHc = 16;      // heads
constexpr int HDc = 128;     // head dim
constexpr int Tc  = 1024;    // sequence length
constexpr int Cc  = 2048;    // channels
constexpr int Bc  = 4;       // batch
constexpr int Mc  = Bc * Tc; // 4096 token rows

typedef __attribute__((ext_vector_type(16))) _Float16 v16h;
typedef __attribute__((ext_vector_type(8)))  _Float16 v8h;
typedef __attribute__((ext_vector_type(4)))  _Float16 v4h;
typedef __attribute__((ext_vector_type(8)))  float    v8f;
typedef __attribute__((ext_vector_type(4)))  unsigned v4u;
typedef __attribute__((ext_vector_type(8)))  unsigned v8u;

__device__ __forceinline__ v16h pack16(v8h lo, v8h hi) {
  v16h r;
#pragma unroll
  for (int t = 0; t < 8; ++t) { r[t] = lo[t]; r[t + 8] = hi[t]; }
  return r;
}

__device__ __forceinline__ v8f wmma_f16(v16h a, v16h b, v8f c) {
  return __builtin_amdgcn_wmma_f32_16x16x32_f16(
      /*neg_a=*/false, a, /*neg_b=*/false, b,
      /*c_mod=*/(short)0, c, /*reuse_a=*/false, /*reuse_b=*/false);
}

// ds_swizzle group-of-32 xor pattern: one instruction, no index VGPR.
#define SWZ_XOR(x, m) \
  __int_as_float(__builtin_amdgcn_ds_swizzle(__float_as_int(x), (((m) << 10) | 0x1f)))

// ---------------------------------------------------------------------------
// Tensor Data Mover: 2-D f16 tile (tile_d0 contiguous elems x tile_d1 rows)
// global -> LDS. D# per CDNA5 ISA:
//  g0: count=1 | lds_addr | global_addr | type=2 ; g1: data_size=1 (2B),
//  workgroup_mask=0 (not in cluster), dims/strides in element units.
//  Groups 2/3 zero: tile_dim2/3 = 0 => unused (2-D tile).
// ---------------------------------------------------------------------------
__device__ __forceinline__ void tdm_load_2d(unsigned lds_off, unsigned long long gaddr,
                                            unsigned tensor_d0, unsigned tensor_d1,
                                            unsigned stride0, unsigned tile_d0,
                                            unsigned tile_d1) {
  v4u g0;
  g0[0] = 1u;                                        // count=1 (user, valid)
  g0[1] = lds_off;                                   // lds_addr (bytes)
  g0[2] = (unsigned)(gaddr & 0xffffffffu);           // global_addr[31:0]
  g0[3] = (unsigned)((gaddr >> 32) & 0x01ffffffu) | (2u << 30);  // [56:32] | type=2
  v8u g1;
  g1[0] = (1u << 16);                                // mask=0, data_size=1 (2 bytes)
  g1[1] = (tensor_d0 & 0xffffu) << 16;               // tensor_dim0[15:0]
  g1[2] = (tensor_d0 >> 16) | ((tensor_d1 & 0xffffu) << 16);
  g1[3] = (tensor_d1 >> 16) | (tile_d0 << 16);       // tile_dim0
  g1[4] = tile_d1;                                   // tile_dim1 (tile_dim2=0)
  g1[5] = stride0;                                   // tensor_dim0_stride[31:0]
  g1[6] = 0;
  g1[7] = 0;
  v4u gz = {0u, 0u, 0u, 0u};
  asm volatile("tensor_load_to_lds %0, %1, %2, %3"
               :: "s"(g0), "s"(g1), "s"(gz), "s"(gz)
               : "memory");
}

// ---------------------------------------------------------------------------
// f32 -> f16 elementwise (count multiple of 4)
// ---------------------------------------------------------------------------
__global__ void snn_cvt(const float* __restrict__ in, _Float16* __restrict__ out, int n4) {
  int i = blockIdx.x * blockDim.x + threadIdx.x;
  if (i >= n4) return;
  float4 f = ((const float4*)in)[i];
  v4h o;
  o[0] = (_Float16)f.x; o[1] = (_Float16)f.y; o[2] = (_Float16)f.z; o[3] = (_Float16)f.w;
  ((v4h*)out)[i] = o;
}

// ---------------------------------------------------------------------------
// C[Mc,Cc] = A[Mc,Cc] (f16) x B[Cc,Cc]^T (f16, row-major weights [N,K]).
// Block: 8 waves; block tile 128x128; wave tile 32x64.
// A/B K-step tiles (128x32 f16, 8KB each) staged into double-buffered LDS by
// the TDM; waves read fragments with ds_load_b128.
// MODE 0: f32 row-major (q,k pre-rope) | MODE 1: f16 [B,NH,HD,T] (v, transposed)
// MODE 2: fused 5-step LIF spike-mean -> f32 row-major (final projection)
// ---------------------------------------------------------------------------
template <int MODE>
__global__ __launch_bounds__(256)
void snn_gemm(const _Float16* __restrict__ A, const _Float16* __restrict__ Bw,
              void* __restrict__ Cout) {
  constexpr int K = Cc, N = Cc;
  __shared__ _Float16 asb[2][128 * 32];   // [row 128][k 32]
  __shared__ _Float16 bsb[2][128 * 32];   // [col 128][k 32]
  const int lane = threadIdx.x & 31;
  const int wid  = threadIdx.x >> 5;
  const int lh = lane & 15, ls = lane >> 4;
  const int wm = (wid & 3) * 32;           // wave row offset in block tile
  const int wn = (wid >> 2) * 64;          // wave col offset in block tile
  const int mblk = blockIdx.x * 128;
  const int nblk = blockIdx.y * 128;

  const unsigned long long ag = (unsigned long long)(size_t)(A  + (size_t)mblk * K);
  const unsigned long long bg = (unsigned long long)(size_t)(Bw + (size_t)nblk * K);
  constexpr int NIT = K / 32;

  if (wid == 0) {  // wave 0 drives the TDM for the whole workgroup
    tdm_load_2d((unsigned)(size_t)&asb[0][0], ag, K, 128, K, 32, 128);
    tdm_load_2d((unsigned)(size_t)&bsb[0][0], bg, K, 128, K, 32, 128);
  }

  v8f acc[2][4] = {};

  for (int it = 0; it < NIT; ++it) {
    if (wid == 0) {
      if (it + 1 < NIT) {
        const int nb = (it + 1) & 1;
        tdm_load_2d((unsigned)(size_t)&asb[nb][0], ag + (size_t)(it + 1) * 64,
                    K, 128, K, 32, 128);
        tdm_load_2d((unsigned)(size_t)&bsb[nb][0], bg + (size_t)(it + 1) * 64,
                    K, 128, K, 32, 128);
        __builtin_amdgcn_s_wait_tensorcnt(2);  // in-order => tile `it` landed
      } else {
        __builtin_amdgcn_s_wait_tensorcnt(0);
      }
    }
    __syncthreads();
    const _Float16* at = &asb[it & 1][0];
    const _Float16* bt = &bsb[it & 1][0];

    v16h a[2], b[4];
#pragma unroll
    for (int i = 0; i < 2; ++i) {
      // A-frag (16x32): lanes 0-15 rows, K=0..7/16..23; lanes 16-31 K=8..15/24..31
      const _Float16* p = at + (wm + i * 16 + lh) * 32 + ls * 8;
      a[i] = pack16(*(const v8h*)p, *(const v8h*)(p + 16));
    }
#pragma unroll
    for (int j = 0; j < 4; ++j) {
      // B-frag (32x16): lane n = column; lanes 0-15 K=0..7/16..23, lanes 16-31 K=8..15/24..31
      const _Float16* p = bt + (wn + j * 16 + lh) * 32 + ls * 8;
      b[j] = pack16(*(const v8h*)p, *(const v8h*)(p + 16));
    }
#pragma unroll
    for (int i = 0; i < 2; ++i)
#pragma unroll
      for (int j = 0; j < 4; ++j)
        acc[i][j] = wmma_f16(a[i], b[j], acc[i][j]);
    __syncthreads();   // all reads done before TDM overwrites this buffer
  }

  // C/D layout: VGPR r, lane l -> row = r + 8*(l>>4), col = l&15
#pragma unroll
  for (int i = 0; i < 2; ++i) {
#pragma unroll
    for (int j = 0; j < 4; ++j) {
#pragma unroll
      for (int r = 0; r < 8; ++r) {
        int row = mblk + wm + i * 16 + r + 8 * ls;
        int col = nblk + wn + j * 16 + lh;
        float val = acc[i][j][r];
        if constexpr (MODE == 0) {
          ((float*)Cout)[(size_t)row * N + col] = val;
        } else if constexpr (MODE == 1) {
          int bb = row / Tc, t = row - bb * Tc;      // Tc constexpr -> shifts
          int h = col >> 7, d = col & (HDc - 1);
          ((_Float16*)Cout)[(((size_t)(bb * NHc + h)) * HDc + d) * Tc + t] = (_Float16)val;
        } else {
          float mem = 0.f, s = 0.f;
#pragma unroll
          for (int st = 0; st < 5; ++st) {
            mem = 0.9f * mem + val;
            float sp = (mem > 1.0f) ? 1.0f : 0.0f;
            mem -= sp;
            s += sp;
          }
          ((float*)Cout)[(size_t)row * N + col] = s * 0.2f;
        }
      }
    }
  }
}

// ---------------------------------------------------------------------------
// rotary + rms_norm. One wave per (token row, head). Writes f16 [B,NH,T,HD].
// ---------------------------------------------------------------------------
__global__ __launch_bounds__(256)
void snn_rope(const float* __restrict__ X, const float* __restrict__ Cb,
              const float* __restrict__ Sb, _Float16* __restrict__ Out) {
  int gw = blockIdx.x * (blockDim.x >> 5) + (threadIdx.x >> 5);
  int lane = threadIdx.x & 31;
  int h = gw & (NHc - 1);
  int row = gw >> 4;
  int bb = row / Tc, t = row - bb * Tc;
  const float* x = X + (size_t)row * Cc + h * HDc;

  float o1[2], o2[2];
  float ss = 0.f;
#pragma unroll
  for (int i = 0; i < 2; ++i) {
    int d = lane * 2 + i;                  // d in [0,64)
    float x1 = x[d], x2 = x[d + 64];
    float c = Cb[t * 64 + d], s = Sb[t * 64 + d];
    o1[i] = x1 * c + x2 * s;
    o2[i] = x2 * c - x1 * s;
    ss += o1[i] * o1[i] + o2[i] * o2[i];
  }
  ss += SWZ_XOR(ss, 1);
  ss += SWZ_XOR(ss, 2);
  ss += SWZ_XOR(ss, 4);
  ss += SWZ_XOR(ss, 8);
  ss += SWZ_XOR(ss, 16);
  float rn = rsqrtf(ss * (1.0f / 128.0f) + 1e-6f);

  _Float16* o = Out + (((size_t)(bb * NHc + h)) * Tc + t) * HDc;
#pragma unroll
  for (int i = 0; i < 2; ++i) {
    int d = lane * 2 + i;
    o[d]      = (_Float16)(o1[i] * rn);
    o[d + 64] = (_Float16)(o2[i] * rn);
  }
}

// ---------------------------------------------------------------------------
// Flash attention. grid = (B*NH, T/128). 8 waves; each wave owns 16 q-rows.
// K/V tiles (shared by all 8 waves) staged into double-buffered LDS by the
// TDM; fragments then read with ds_load_b128.
// Q,K: f16 [B,NH,T,HD]; V: f16 [B,NH,HD,T]; Y: f16 [Mc, Cc]
// ---------------------------------------------------------------------------
__global__ __launch_bounds__(256)
void snn_attn(const _Float16* __restrict__ Q, const _Float16* __restrict__ Kh,
              const _Float16* __restrict__ Vt, _Float16* __restrict__ Y) {
  __shared__ _Float16 kbuf[2][32 * HDc];   // [j 32][d 128], 8KB per buffer
  __shared__ _Float16 vbuf[2][HDc * 32];   // [d 128][j 32], 8KB per buffer
  __shared__ _Float16 pbuf[8 * 16 * 40];   // per-wave P tile, stride 40
  const int bh = blockIdx.x;
  const int wid = threadIdx.x >> 5, lane = threadIdx.x & 31;
  const int lh = lane & 15, ls = lane >> 4;
  const int q0 = blockIdx.y * 128 + wid * 16;
  const _Float16* q = Q + (size_t)bh * Tc * HDc;
  _Float16* lp = pbuf + wid * 640;

  // Resident Q fragments (4 K-steps over d)
  v16h qa[4];
#pragma unroll
  for (int kk = 0; kk < 4; ++kk) {
    const _Float16* p = q + (size_t)(q0 + lh) * HDc + kk * 32 + ls * 8;
    qa[kk] = pack16(*(const v8h*)p, *(const v8h*)(p + 16));
  }

  const unsigned long long kg = (unsigned long long)(size_t)(Kh + (size_t)bh * Tc * HDc);
  const unsigned long long vg = (unsigned long long)(size_t)(Vt + (size_t)bh * HDc * Tc);
  constexpr int NIT = Tc / 32;

  if (wid == 0) {  // wave 0 drives the TDM for the whole workgroup
    tdm_load_2d((unsigned)(size_t)&kbuf[0][0], kg, HDc, Tc, HDc, HDc, 32);
    tdm_load_2d((unsigned)(size_t)&vbuf[0][0], vg, Tc, HDc, Tc, 32, HDc);
  }

  v8f acc[8] = {};
  float mrow[8], lrow[8];
#pragma unroll
  for (int i = 0; i < 8; ++i) { mrow[i] = -1e30f; lrow[i] = 0.f; }
  const float scale = 0.08838834764831845f;    // 1/sqrt(128)

  for (int it = 0; it < NIT; ++it) {
    if (wid == 0) {
      if (it + 1 < NIT) {
        const int nb = (it + 1) & 1;
        tdm_load_2d((unsigned)(size_t)&kbuf[nb][0],
                    kg + (size_t)(it + 1) * 32 * HDc * 2, HDc, Tc, HDc, HDc, 32);
        tdm_load_2d((unsigned)(size_t)&vbuf[nb][0],
                    vg + (size_t)(it + 1) * 32 * 2, Tc, HDc, Tc, 32, HDc);
        __builtin_amdgcn_s_wait_tensorcnt(2);  // in-order => tile `it` landed
      } else {
        __builtin_amdgcn_s_wait_tensorcnt(0);
      }
    }
    __syncthreads();                            // tile `it` visible to all waves
    const _Float16* kt = &kbuf[it & 1][0];
    const _Float16* vl = &vbuf[it & 1][0];

    // S tiles for keys j0..j0+15 and j0+16..j0+31 (from LDS)
    v8f s2[2];
#pragma unroll
    for (int jt = 0; jt < 2; ++jt) {
      v8f sv = {};
#pragma unroll
      for (int kk = 0; kk < 4; ++kk) {
        const _Float16* p = kt + (jt * 16 + lh) * HDc + kk * 32 + ls * 8;
        v16h bf = pack16(*(const v8h*)p, *(const v8h*)(p + 16));
        sv = wmma_f16(qa[kk], bf, sv);
      }
      s2[jt] = sv;
    }
    // Online softmax: row r = i + 8*(lane>>4) lives across a 16-lane group
#pragma unroll
    for (int i = 0; i < 8; ++i) {
      float e0 = s2[0][i] * scale;
      float e1 = s2[1][i] * scale;
      float m = fmaxf(e0, e1);
      m = fmaxf(m, SWZ_XOR(m, 1));
      m = fmaxf(m, SWZ_XOR(m, 2));
      m = fmaxf(m, SWZ_XOR(m, 4));
      m = fmaxf(m, SWZ_XOR(m, 8));
      float mnew = fmaxf(mrow[i], m);
      float corr = __expf(mrow[i] - mnew);
      float p0 = __expf(e0 - mnew);
      float p1 = __expf(e1 - mnew);
      float ps = p0 + p1;
      ps += SWZ_XOR(ps, 1);
      ps += SWZ_XOR(ps, 2);
      ps += SWZ_XOR(ps, 4);
      ps += SWZ_XOR(ps, 8);
      lrow[i] = lrow[i] * corr + ps;
      mrow[i] = mnew;
#pragma unroll
      for (int n = 0; n < 8; ++n) acc[n][i] *= corr;
      lp[(i + 8 * ls) * 40 + lh]      = (_Float16)p0;
      lp[(i + 8 * ls) * 40 + 16 + lh] = (_Float16)p1;
    }
    // Reload P as A-fragment (same-wave LDS ops are in-order)
    const _Float16* pp = lp + lh * 40 + ls * 8;
    v16h pa = pack16(*(const v8h*)pp, *(const v8h*)(pp + 16));
    // P @ V from LDS V tile
#pragma unroll
    for (int n = 0; n < 8; ++n) {
      const _Float16* p = vl + (n * 16 + lh) * 32 + ls * 8;
      v16h bf = pack16(*(const v8h*)p, *(const v8h*)(p + 16));
      acc[n] = wmma_f16(pa, bf, acc[n]);
    }
    __syncthreads();   // all reads of buffer `it` done before TDM overwrites it
  }

  const int bb = bh >> 4, h = bh & (NHc - 1);
  float inv[8];
#pragma unroll
  for (int i = 0; i < 8; ++i) inv[i] = 1.0f / lrow[i];
#pragma unroll
  for (int n = 0; n < 8; ++n)
#pragma unroll
    for (int i = 0; i < 8; ++i) {
      int row = q0 + i + 8 * ls;
      int col = h * HDc + n * 16 + lh;
      Y[((size_t)(bb * Tc) + row) * Cc + col] = (_Float16)(acc[n][i] * inv[i]);
    }
}

// ---------------------------------------------------------------------------
extern "C" void kernel_launch(void* const* d_in, const int* in_sizes, int n_in,
                              void* d_out, int out_size, void* d_ws, size_t ws_size,
                              hipStream_t stream) {
  const float* x  = (const float*)d_in[0];
  const float* cb = (const float*)d_in[1];
  const float* sb = (const float*)d_in[2];
  const float* W[4] = {(const float*)d_in[3], (const float*)d_in[4],
                       (const float*)d_in[5], (const float*)d_in[6]};

  char* ws = (char*)d_ws;
  auto alloc = [&](size_t bytes) -> void* {
    void* p = (void*)ws;
    ws += (bytes + 255) & ~(size_t)255;
    return p;
  };
  _Float16* xh = (_Float16*)alloc((size_t)Mc * Cc * 2);
  _Float16* Wh[4];
  for (int i = 0; i < 4; ++i) Wh[i] = (_Float16*)alloc((size_t)Cc * Cc * 2);
  float*    qf = (float*)   alloc((size_t)Mc * Cc * 4);
  float*    kf = (float*)   alloc((size_t)Mc * Cc * 4);
  _Float16* qh = (_Float16*)alloc((size_t)Mc * Cc * 2);
  _Float16* kh = (_Float16*)alloc((size_t)Mc * Cc * 2);
  _Float16* vt = (_Float16*)alloc((size_t)Mc * Cc * 2);
  _Float16* yh = (_Float16*)alloc((size_t)Mc * Cc * 2);

  int n4 = Mc * Cc / 4;
  snn_cvt<<<(n4 + 255) / 256, 256, 0, stream>>>(x, xh, n4);
  n4 = Cc * Cc / 4;
  for (int i = 0; i < 4; ++i)
    snn_cvt<<<(n4 + 255) / 256, 256, 0, stream>>>(W[i], Wh[i], n4);

  dim3 gg(Mc / 128, Cc / 128);
  snn_gemm<0><<<gg, 256, 0, stream>>>(xh, Wh[0], qf);
  snn_gemm<0><<<gg, 256, 0, stream>>>(xh, Wh[1], kf);
  snn_gemm<1><<<gg, 256, 0, stream>>>(xh, Wh[2], vt);

  int blocks = Mc * NHc / 8;   // one wave per (token, head)
  snn_rope<<<blocks, 256, 0, stream>>>(qf, cb, sb, qh);
  snn_rope<<<blocks, 256, 0, stream>>>(kf, cb, sb, kh);

  snn_attn<<<dim3(Bc * NHc, Tc / 128), 256, 0, stream>>>(qh, kh, vt, yh);

  snn_gemm<2><<<gg, 256, 0, stream>>>(yh, Wh[3], (float*)d_out);
}
